// RNN_86148454023838
// MI455X (gfx1250) — matrix operand order repaired and checked by reading, not compiled, not run on hardware
//
#include <hip/hip_runtime.h>
#include <hip/hip_bf16.h>

typedef __attribute__((ext_vector_type(16))) __bf16 v16bf;
typedef __attribute__((ext_vector_type(8)))  float  v8f;

#define Bsz   64
#define Tsz   2048
#define Nsz   512
#define ALPHAv 0.1f
// noise_std = sqrt(2 * 0.1 * 0.05^2)
#define NSTD  0.022360679774997897f

// LDS layout (bytes)
#define LDS_STATE  0                    // 16*512 f32            = 32768
#define LDS_FRAG   32768                // 2 * 16*32*16 bf16     = 32768
#define LDS_U      65536                // 2 * 32 f32            = 256
#define LDS_NOISE  65792                // 2 * 16*512 f32        = 65536
#define LDS_TOTAL  131328

// -----------------------------------------------------------------------------
// Pack W_rec (N x N, f32, row-major) into bf16 in WMMA B-operand fragment order.
//   B chunk = 32x16 (K x N):  lanes 0-15: N=lane, K=(e&7)+16*(e>>3) ; lanes 16-31: N=lane-16, K=8+(e&7)+16*(e>>3)
//   ws index: ((w*16 + c)*32 + lane)*16 + e
// -----------------------------------------------------------------------------
__global__ void pack_wrec_bf16(const float* __restrict__ Wrec,
                               __bf16* __restrict__ wsbf) {
  int idx = blockIdx.x * blockDim.x + threadIdx.x;   // N*N threads
  int e = idx & 15;
  int L = (idx >> 4) & 31;
  int c = (idx >> 9) & 15;
  int w = (idx >> 13) & 31;
  int k = 32 * c + (e & 7) + 8 * (L >> 4) + 16 * (e >> 3);
  int n = 16 * w + (L & 15);
  wsbf[idx] = (__bf16)Wrec[n * Nsz + k];
}

// -----------------------------------------------------------------------------
// Persistent recurrent scan.  grid = 4 (batch tiles of 16), block = 1024 (32 waves).
// Wave w owns D tile rows [16*blk .. +16) x cols [16w .. +16).
// -----------------------------------------------------------------------------
__global__ void rnn_scan(const float* __restrict__ u,
                         const float* __restrict__ init_state,
                         const float* __restrict__ noise,
                         const __bf16* __restrict__ wsbf,
                         const float* __restrict__ W_in,
                         float* __restrict__ states,
                         float* __restrict__ act) {
  extern __shared__ char smem_raw[];
  float*  s_state = (float*)(smem_raw + LDS_STATE);
  __bf16* s_frag  = (__bf16*)(smem_raw + LDS_FRAG);
  float*  s_u     = (float*)(smem_raw + LDS_U);
  float*  s_noise = (float*)(smem_raw + LDS_NOISE);

  const int tid   = threadIdx.x;
  const int lane  = tid & 31;
  const int wave  = tid >> 5;          // 0..31 : n-tile
  const int lhalf = lane >> 4;
  const int Nl    = lane & 15;
  const int n     = (wave << 4) | Nl;  // column owned in C/D
  const int b0    = blockIdx.x * 16;   // batch base

  // frag-scatter constants for column n (A-operand layout, K = n next step)
  const int c_t  = n >> 5;
  const int kin  = n & 31;
  const int e_t  = (kin & 7) | (((kin >> 4) & 1) << 3);
  const int kbit = (kin >> 3) & 1;

  const float wi0 = W_in[n * 2 + 0];
  const float wi1 = W_in[n * 2 + 1];

  // ---- async noise staging constants (wave pair covers row nm = wave/2) ----
  const int      nm       = wave >> 1;
  const unsigned ncolB    = ((wave & 1) << 8) + (unsigned)lane * 4u;   // float col
  const unsigned nz_voff  = (nm * (unsigned)(Tsz * Nsz) + ncolB) * 4u; // global byte off
  const unsigned nz_lds0  = LDS_NOISE + (nm * 512u + ncolB) * 4u;      // LDS byte addr

  // ---- prologue: async-copy noise[:, 0, :] into LDS buffer 0 ----
  {
    const float* sbase = noise + ((size_t)b0 * Tsz + 0) * Nsz;
    asm volatile("global_load_async_to_lds_b128 %0, %1, %2 offset:0"
                 :: "v"(nz_lds0), "v"(nz_voff), "s"(sbase) : "memory");
    asm volatile("global_load_async_to_lds_b128 %0, %1, %2 offset:512"
                 :: "v"(nz_lds0), "v"(nz_voff), "s"(sbase) : "memory");
  }

  // ---- t = 0: init state, emit states/act row 0, seed frag buffer 0 ----
  for (int i = tid; i < 16 * Nsz; i += 1024) {
    int m  = i >> 9;
    int nn = i & (Nsz - 1);
    float s = init_state[(b0 + m) * Nsz + nn];
    s_state[i] = s;
    float r = fmaxf(s, 0.f);
    int g = ((b0 + m) * Tsz + 0) * Nsz + nn;
    states[g] = s;
    act[g]    = r;
    int cc = nn >> 5, ki = nn & 31;
    int ee = (ki & 7) | (((ki >> 4) & 1) << 3);
    int lt = m + 16 * ((ki >> 3) & 1);
    s_frag[(cc * 32 + lt) * 16 + ee] = (__bf16)r;
  }
  if (tid < 32) {  // u[:, 0, :]
    int m = tid >> 1, j = tid & 1;
    s_u[m * 2 + j] = u[((b0 + m) * Tsz + 0) * 2 + j];
  }
  asm volatile("s_wait_asynccnt 0x0" ::: "memory");
  __syncthreads();

  int cur = 0;
  for (int t = 1; t < Tsz; ++t) {
    const int nxt = cur ^ 1;

    // ---- kick off async copy of noise[:, t, :] into buffer `nxt` ----
    {
      unsigned ldsA = nz_lds0 + (unsigned)nxt * 32768u;
      const float* sbase = noise + ((size_t)b0 * Tsz + t) * Nsz;
      asm volatile("global_load_async_to_lds_b128 %0, %1, %2 offset:0"
                   :: "v"(ldsA), "v"(nz_voff), "s"(sbase) : "memory");
      asm volatile("global_load_async_to_lds_b128 %0, %1, %2 offset:512"
                   :: "v"(ldsA), "v"(nz_voff), "s"(sbase) : "memory");
    }

    // Opaque scalar zero: prevents LICM from hoisting the 16 loop-invariant
    // B-fragment loads into VGPRs (spills), while keeping the loads attached
    // to the global `wsbf` pointer so they lower as global_load (not flat).
    unsigned woff = 0;
    asm volatile("" : "+s"(woff));
    const __bf16* wbp = wsbf + (wave * 16) * 512 + (unsigned)lane * 16 + woff;

    const __bf16* fr = s_frag + cur * (16 * 32 * 16);
    const float*  uu = s_u + cur * 32;

    // ---- relu(s_{t-1}) @ W_rec^T over K = 512 : 16 WMMAs, 2 accumulators ----
    v8f acc0 = {}; v8f acc1 = {};
    #pragma unroll
    for (int c = 0; c < 16; c += 2) {
      v16bf a0  = *(const v16bf*)(fr + (c * 32 + lane) * 16);
      v16bf b0v = *(const v16bf*)(wbp + c * 512);
      acc0 = __builtin_amdgcn_wmma_f32_16x16x32_bf16(false, a0, false, b0v,
                                                     (short)0, acc0, false, false);
      v16bf a1  = *(const v16bf*)(fr + ((c + 1) * 32 + lane) * 16);
      v16bf b1v = *(const v16bf*)(wbp + (c + 1) * 512);
      acc1 = __builtin_amdgcn_wmma_f32_16x16x32_bf16(false, a1, false, b1v,
                                                     (short)0, acc1, false, false);
    }
    v8f acc = acc0 + acc1;

    // ---- state update: s_t = (1-a)s + a(acc + u@W_in^T) + nstd*noise_{t-1} ----
    const float* nzbuf = s_noise + cur * (16 * 512);
    __bf16* fw = s_frag + nxt * (16 * 32 * 16) + (c_t * 32) * 16 + e_t;
    #pragma unroll
    for (int v = 0; v < 8; ++v) {
      int M = v + 8 * lhalf;                     // D layout: M = vgpr + 8*(lane/16)
      float s_old = s_state[M * Nsz + n];
      float drive = uu[M * 2 + 0] * wi0 + uu[M * 2 + 1] * wi1;
      float nz    = nzbuf[M * 512 + n];          // noise[b, t-1, n] (prefetched)
      float s_new = (1.f - ALPHAv) * s_old + ALPHAv * (acc[v] + drive) + NSTD * nz;
      s_state[M * Nsz + n] = s_new;
      int g = ((b0 + M) * Tsz + t) * Nsz + n;
      states[g] = s_new;
      float r = fmaxf(s_new, 0.f);
      act[g] = r;
      int lane_t = M + 16 * kbit;                // scatter into next A-frag buffer
      fw[lane_t * 16] = (__bf16)r;
    }

    if (tid < 32) {  // stage u[:, t, :] for next iteration
      int m = tid >> 1, j = tid & 1;
      s_u[nxt * 32 + m * 2 + j] = u[((b0 + m) * Tsz + t) * 2 + j];
    }

    asm volatile("s_wait_asynccnt 0x0" ::: "memory");
    __syncthreads();
    cur = nxt;
  }
}

// -----------------------------------------------------------------------------
// out[b,t,:] = act[b,t,:] @ W_out^T   (one wave per (b,t) row, shuffle reduce)
// -----------------------------------------------------------------------------
__global__ void out_proj(const float* __restrict__ act,
                         const float* __restrict__ Wout,
                         float* __restrict__ out) {
  int row  = (blockIdx.x * blockDim.x + threadIdx.x) >> 5;
  int lane = threadIdx.x & 31;
  if (row >= Bsz * Tsz) return;
  const float* a = act + (size_t)row * Nsz;
  float d0 = 0.f, d1 = 0.f;
  #pragma unroll
  for (int i = 0; i < 16; ++i) {
    int nn = lane + 32 * i;
    float x = a[nn];
    d0 += x * Wout[nn];
    d1 += x * Wout[Nsz + nn];
  }
  #pragma unroll
  for (int off = 16; off > 0; off >>= 1) {
    d0 += __shfl_down(d0, off, 32);
    d1 += __shfl_down(d1, off, 32);
  }
  if (lane == 0) { out[row * 2 + 0] = d0; out[row * 2 + 1] = d1; }
}

extern "C" void kernel_launch(void* const* d_in, const int* in_sizes, int n_in,
                              void* d_out, int out_size, void* d_ws, size_t ws_size,
                              hipStream_t stream) {
  const float* u          = (const float*)d_in[0];
  const float* init_state = (const float*)d_in[1];
  const float* noise      = (const float*)d_in[2];
  const float* W_rec      = (const float*)d_in[3];
  const float* W_in       = (const float*)d_in[4];
  const float* W_out      = (const float*)d_in[5];

  float* out    = (float*)d_out;                         // (B,T,2)
  float* act    = out + Bsz * Tsz * 2;                   // (B,T,N)
  float* states = act + (size_t)Bsz * Tsz * Nsz;         // (B,T,N)
  __bf16* wsbf  = (__bf16*)d_ws;                         // N*N bf16 = 512KB scratch

  pack_wrec_bf16<<<(Nsz * Nsz) / 256, 256, 0, stream>>>(W_rec, wsbf);

  rnn_scan<<<4, 1024, LDS_TOTAL, stream>>>(u, init_state, noise, wsbf, W_in,
                                           states, act);

  out_proj<<<(Bsz * Tsz * 32) / 256, 256, 0, stream>>>(act, W_out, out);
}
